// GATModel_79422535238234
// MI455X (gfx1250) — compile-verified, matrix operand order checked by reading, not run on hardware
//
#include <hip/hip_runtime.h>
#include <math.h>

// ---------------------------------------------------------------------------
// Feature detection: CDNA5 async global->LDS copy (ASYNCcnt-tracked).
// ---------------------------------------------------------------------------
#if defined(__has_builtin)
#  if __has_builtin(__builtin_amdgcn_global_load_async_to_lds_b128)
#    define GAT_ASYNC_LDS 1
#  endif
#endif
#ifndef GAT_ASYNC_LDS
#  define GAT_ASYNC_LDS 0
#endif

// ---------------------------------------------------------------------------
// Types for CDNA5 WMMA (wave32): bf16 A/B fragments (16 bf16 = 8 VGPRs),
// f32 accumulator (8 VGPRs).
// ---------------------------------------------------------------------------
typedef __attribute__((ext_vector_type(16))) __bf16 v16bf;
typedef __attribute__((ext_vector_type(8)))  float  v8f;

// Exact operand type of the async-LDS builtin (from the compiler diagnostic):
// v4i in AS1 (printed "__device__") and AS3 (printed "__shared__").
typedef int v4i __attribute__((vector_size(16)));
typedef __attribute__((address_space(1))) v4i* as1_v4i;
typedef __attribute__((address_space(3))) v4i* as3_v4i;

union FragBF {
    v16bf v;
    uint4 q[2];
};

__device__ __forceinline__ unsigned short f2bf(float f) {
    unsigned int u = __float_as_uint(f);
    unsigned int r = (u + 0x7FFFu + ((u >> 16) & 1u)) >> 16;  // RNE
    return (unsigned short)r;
}

// Copy 32 bytes (16 bf16) from global to LDS. Async path: two
// global_load_async_to_lds_b128 (imm offset applies to both addresses).
__device__ __forceinline__ void stage_tile(const unsigned short* gp,
                                           unsigned short* lp) {
#if GAT_ASYNC_LDS
    // ptrtoint -> inttoptr: AS1 keeps the 64-bit global address; AS3 truncates
    // the generic LDS address to its 32-bit wave-relative offset (aperture rule).
    __builtin_amdgcn_global_load_async_to_lds_b128(
        (as1_v4i)(unsigned long long)gp,
        (as3_v4i)(unsigned long long)lp, 0, 0);
    __builtin_amdgcn_global_load_async_to_lds_b128(
        (as1_v4i)(unsigned long long)gp,
        (as3_v4i)(unsigned long long)lp, 16, 0);
#else
    const uint4* p = reinterpret_cast<const uint4*>(gp);
    uint4 d0 = p[0];
    uint4 d1 = p[1];
    uint4* q = reinterpret_cast<uint4*>(lp);
    q[0] = d0;
    q[1] = d1;
#endif
}

__device__ __forceinline__ void wait_async_zero() {
#if GAT_ASYNC_LDS
#  if __has_builtin(__builtin_amdgcn_s_wait_asynccnt)
    __builtin_amdgcn_s_wait_asynccnt(0);
#  else
    asm volatile("s_wait_asynccnt 0x0" ::: "memory");
#  endif
#endif
}

// ---------------------------------------------------------------------------
// fp32 -> bf16 conversion
// ---------------------------------------------------------------------------
__global__ void cvt_bf16_kernel(const float* __restrict__ in,
                                unsigned short* __restrict__ out,
                                long long n) {
    long long i = (long long)blockIdx.x * blockDim.x + threadIdx.x;
    if (i < n) out[i] = f2bf(in[i]);
}

// W [K x M] fp32 -> Wt [M x K] bf16
__global__ void transpose_w_kernel(const float* __restrict__ W,
                                   unsigned short* __restrict__ Wt,
                                   int K, int M) {
    int i = blockIdx.x * blockDim.x + threadIdx.x;
    if (i < K * M) {
        int k = i / M, m = i % M;
        Wt[m * K + k] = f2bf(W[i]);
    }
}

__global__ void fill_kernel(float* __restrict__ p, float v, long long n) {
    long long i = (long long)blockIdx.x * blockDim.x + threadIdx.x;
    if (i < n) p[i] = v;
}

// ---------------------------------------------------------------------------
// GEMM: C[N x M] = A[N x K] * B[K x M]; A bf16 row-major, B supplied
// transposed (Bt [M x K] bf16), fp32 accumulate via v_wmma_f32_16x16x32_bf16.
// Block = 256 threads = 8 waves; block tile 128x128; wave tile 32x64.
// A tiles are double-buffered in LDS and streamed with async global->LDS
// copies (ASYNCcnt); B fragments come straight from the L2-resident Bt.
//
// bf16 A-fragment layout (ISA 7.12.2): lane holds row (lane%16); element j of
// the v16 holds K = 16*(j/8) + 8*(lane/16) + (j%8) -> two contiguous 16B
// chunks per lane. B mirrors this with lane%16 = output column.
// ---------------------------------------------------------------------------
__global__ __launch_bounds__(256) void gemm_wmma_kernel(
    const unsigned short* __restrict__ A,   // N x K bf16
    const unsigned short* __restrict__ Bt,  // M x K bf16
    float* __restrict__ C,                  // N x M f32
    int N, int K, int M) {
    __shared__ unsigned short As[2][128][40];  // 32 K + 8 pad (16B-aligned rows)

    const int tid  = threadIdx.x;
    const int lane = tid & 31;
    const int wave = tid >> 5;
    const int wr   = wave & 3;   // 4 waves -> 128 rows
    const int wc   = wave >> 2;  // 2 waves -> 128 cols
    const int row0 = blockIdx.x * 128;
    const int col0 = blockIdx.y * 128;
    const int lrow = lane & 15;
    const int bsel = (lane >> 4) << 3;  // 0 or 8 (K sub-chunk select)

    // --- staging indices (clamped: OOB rows fetch row N-1; outputs guarded) ---
    const int sr  = tid >> 1;
    const int skc = (tid & 1) << 4;  // 0 or 16
    int grow = row0 + sr;
    if (grow >= N) grow = N - 1;
    const unsigned short* gp0 = A + (long long)grow * K + skc;

    // --- clamped B columns (OOB cols read finite junk; stores guarded) ---
    int bcol[4];
    for (int cf = 0; cf < 4; ++cf) {
        int col = col0 + wc * 64 + cf * 16 + lrow;
        bcol[cf] = col < M ? col : M - 1;
    }

    v8f acc[2][4];
    for (int r = 0; r < 2; ++r)
        for (int c = 0; c < 4; ++c)
            acc[r][c] = (v8f){0.f, 0.f, 0.f, 0.f, 0.f, 0.f, 0.f, 0.f};

    // prologue: stream tile k0=0 into buffer 0
    stage_tile(gp0, &As[0][sr][skc]);
    int buf = 0;

    for (int k0 = 0; k0 < K; k0 += 32) {
        wait_async_zero();   // my async writes for As[buf] are in LDS
        __syncthreads();     // everyone's writes visible; prev reads of buf^1 done
        if (k0 + 32 < K)     // stream next tile into the other buffer
            stage_tile(gp0 + k0 + 32, &As[buf ^ 1][sr][skc]);

        // --- A fragments from LDS (two 16B chunks per lane) ---
        FragBF a[2];
        for (int r = 0; r < 2; ++r) {
            int arow = wr * 32 + r * 16 + lrow;
            a[r].q[0] = *reinterpret_cast<const uint4*>(&As[buf][arow][bsel]);
            a[r].q[1] = *reinterpret_cast<const uint4*>(&As[buf][arow][16 + bsel]);
        }

        // --- B fragments from Bt, then 8 WMMAs per wave ---
        for (int cf = 0; cf < 4; ++cf) {
            const uint4* p =
                reinterpret_cast<const uint4*>(Bt + (long long)bcol[cf] * K + k0);
            FragBF b;
            b.q[0] = p[bsel >> 3];
            b.q[1] = p[2 + (bsel >> 3)];
            for (int r = 0; r < 2; ++r) {
                acc[r][cf] = __builtin_amdgcn_wmma_f32_16x16x32_bf16(
                    false, a[r].v, false, b.v, (short)0, acc[r][cf], false, false);
            }
        }
        buf ^= 1;
    }

    // --- store: lane l, vgpr v -> row = base + v + 8*(l/16), col = l%16 ---
    for (int r = 0; r < 2; ++r) {
        int mrow0 = row0 + wr * 32 + r * 16 + ((lane >> 4) << 3);
        for (int cf = 0; cf < 4; ++cf) {
            int col = col0 + wc * 64 + cf * 16 + lrow;
            if (col < M) {
                for (int v = 0; v < 8; ++v) {
                    int rr = mrow0 + v;
                    if (rr < N) C[(long long)rr * M + col] = acc[r][cf][v];
                }
            }
        }
    }
}

// ---------------------------------------------------------------------------
// Attention coefficients: one warp per (node, head), C = 64 fixed.
// ---------------------------------------------------------------------------
__global__ void attn_coef_kernel(const float* __restrict__ xw,
                                 const float* __restrict__ att_src,
                                 const float* __restrict__ att_dst,
                                 float* __restrict__ as_out,
                                 float* __restrict__ ad_out,
                                 int N, int H) {
    long long w = ((long long)blockIdx.x * blockDim.x + threadIdx.x) >> 5;
    int lane = threadIdx.x & 31;
    if (w >= (long long)N * H) return;
    int n = (int)(w / H), h = (int)(w % H);
    const float* xp  = xw + ((long long)n * H + h) * 64;
    const float* asp = att_src + h * 64;
    const float* adp = att_dst + h * 64;
    float x0 = xp[lane], x1 = xp[lane + 32];
    float vs = x0 * asp[lane] + x1 * asp[lane + 32];
    float vd = x0 * adp[lane] + x1 * adp[lane + 32];
    for (int off = 16; off > 0; off >>= 1) {
        vs += __shfl_down(vs, off, 32);
        vd += __shfl_down(vd, off, 32);
    }
    if (lane == 0) {
        as_out[w] = vs;
        ad_out[w] = vd;
    }
}

__device__ __forceinline__ float leaky02(float v) {
    return v > 0.f ? v : 0.2f * v;
}

__device__ __forceinline__ void edge_nodes(const int* __restrict__ ei, int e,
                                           int E, int& s, int& d) {
    if (e < E) {
        s = ei[e];
        d = ei[E + e];
    } else {
        s = d = e - E;  // appended self-loop
    }
}

// Pass 1: per-destination running max (ordered int/uint float atomic-max).
__global__ void edge_max_kernel(const int* __restrict__ ei,
                                const float* __restrict__ as_,
                                const float* __restrict__ ad_,
                                float* __restrict__ m,
                                int E, int N, int H) {
    long long idx = (long long)blockIdx.x * blockDim.x + threadIdx.x;
    long long tot = (long long)(E + N) * H;
    if (idx >= tot) return;
    int e = (int)(idx / H), h = (int)(idx % H);
    int s, d;
    edge_nodes(ei, e, E, s, d);
    float v = leaky02(as_[(long long)s * H + h] + ad_[(long long)d * H + h]);
    float* addr = m + (long long)d * H + h;
    int iv = __float_as_int(v);
    if (iv >= 0)
        atomicMax((int*)addr, iv);
    else
        atomicMin((unsigned int*)addr, (unsigned int)iv);
}

// Pass 2: sum of exp(e - m[dst]).
__global__ void edge_sum_kernel(const int* __restrict__ ei,
                                const float* __restrict__ as_,
                                const float* __restrict__ ad_,
                                const float* __restrict__ m,
                                float* __restrict__ ssum,
                                int E, int N, int H) {
    long long idx = (long long)blockIdx.x * blockDim.x + threadIdx.x;
    long long tot = (long long)(E + N) * H;
    if (idx >= tot) return;
    int e = (int)(idx / H), h = (int)(idx % H);
    int s, d;
    edge_nodes(ei, e, E, s, d);
    float v  = leaky02(as_[(long long)s * H + h] + ad_[(long long)d * H + h]);
    float ex = __expf(v - m[(long long)d * H + h]);
    atomicAdd(&ssum[(long long)d * H + h], ex);
}

// Pass 3: out[dst,h,:] += alpha * xw[src,h,:], one warp per (edge, head).
__global__ void edge_aggr_kernel(const int* __restrict__ ei,
                                 const float* __restrict__ as_,
                                 const float* __restrict__ ad_,
                                 const float* __restrict__ m,
                                 const float* __restrict__ ssum,
                                 const float* __restrict__ xw,
                                 float* __restrict__ out,
                                 int E, int N, int H) {
    long long w = ((long long)blockIdx.x * blockDim.x + threadIdx.x) >> 5;
    int lane = threadIdx.x & 31;
    long long tot = (long long)(E + N) * H;
    if (w >= tot) return;
    int e = (int)(w / H), h = (int)(w % H);
    int s, d;
    edge_nodes(ei, e, E, s, d);
    float v     = leaky02(as_[(long long)s * H + h] + ad_[(long long)d * H + h]);
    float alpha = __expf(v - m[(long long)d * H + h]) / ssum[(long long)d * H + h];
    const float* xp = xw + ((long long)s * H + h) * 64;
    float*       op = out + ((long long)d * H + h) * 64;
    atomicAdd(op + lane,      alpha * xp[lane]);
    atomicAdd(op + lane + 32, alpha * xp[lane + 32]);
}

// bias add (+ optional ELU)
__global__ void bias_act_kernel(float* __restrict__ h,
                                const float* __restrict__ bias,
                                long long n, int M, int do_elu) {
    long long i = (long long)blockIdx.x * blockDim.x + threadIdx.x;
    if (i >= n) return;
    float v = h[i] + bias[(int)(i % M)];
    if (do_elu) v = v > 0.f ? v : (__expf(v) - 1.f);
    h[i] = v;
}

// column sums of h3 [N x 64] -> g[64] (g pre-zeroed)
__global__ void colsum_kernel(const float* __restrict__ h3,
                              float* __restrict__ g, int N) {
    __shared__ float acc[64];
    if (threadIdx.x < 64) acc[threadIdx.x] = 0.f;
    __syncthreads();
    int col     = threadIdx.x & 63;
    int rstart  = blockIdx.x * (blockDim.x >> 6) + (threadIdx.x >> 6);
    int rstride = gridDim.x * (blockDim.x >> 6);
    float p = 0.f;
    for (int r = rstart; r < N; r += rstride)
        p += h3[(long long)r * 64 + col];
    atomicAdd(&acc[col], p);
    __syncthreads();
    if (threadIdx.x < 64) atomicAdd(&g[threadIdx.x], acc[threadIdx.x]);
}

// out[1 x OUT] = (g/N) @ Wc + bc
__global__ void final_kernel(const float* __restrict__ g,
                             const float* __restrict__ Wc,
                             const float* __restrict__ bc,
                             float* __restrict__ out, int N, int OUT) {
    int j = threadIdx.x;
    if (j >= OUT) return;
    float inv = 1.f / (float)N;
    float sum = 0.f;
    for (int c = 0; c < 64; ++c) sum += (g[c] * inv) * Wc[c * OUT + j];
    out[j] = sum + bc[j];
}

// ---------------------------------------------------------------------------
// Host orchestration
// ---------------------------------------------------------------------------
static void run_gat_layer(const unsigned short* act_bf, int K, int M, int H,
                          const float* W, const float* att_s, const float* att_d,
                          const float* bias, int do_elu,
                          const int* ei, int N, int E,
                          unsigned short* Wt, float* xw, float* outb,
                          float* a_src, float* a_dst, float* mbuf, float* sbuf,
                          hipStream_t stream) {
    int KM = K * M;
    transpose_w_kernel<<<(KM + 255) / 256, 256, 0, stream>>>(W, Wt, K, M);

    dim3 ggrid((N + 127) / 128, (M + 127) / 128);
    gemm_wmma_kernel<<<ggrid, 256, 0, stream>>>(act_bf, Wt, xw, N, K, M);

    long long NH = (long long)N * H;
    attn_coef_kernel<<<(unsigned)((NH * 32 + 255) / 256), 256, 0, stream>>>(
        xw, att_s, att_d, a_src, a_dst, N, H);

    fill_kernel<<<(unsigned)((NH + 255) / 256), 256, 0, stream>>>(mbuf, -INFINITY, NH);
    fill_kernel<<<(unsigned)((NH + 255) / 256), 256, 0, stream>>>(sbuf, 0.f, NH);
    long long NM = (long long)N * M;
    fill_kernel<<<(unsigned)((NM + 255) / 256), 256, 0, stream>>>(outb, 0.f, NM);

    long long totE = (long long)(E + N) * H;
    edge_max_kernel<<<(unsigned)((totE + 255) / 256), 256, 0, stream>>>(
        ei, a_src, a_dst, mbuf, E, N, H);
    edge_sum_kernel<<<(unsigned)((totE + 255) / 256), 256, 0, stream>>>(
        ei, a_src, a_dst, mbuf, sbuf, E, N, H);
    edge_aggr_kernel<<<(unsigned)((totE * 32 + 255) / 256), 256, 0, stream>>>(
        ei, a_src, a_dst, mbuf, sbuf, xw, outb, E, N, H);

    bias_act_kernel<<<(unsigned)((NM + 255) / 256), 256, 0, stream>>>(
        outb, bias, NM, M, do_elu);
}

extern "C" void kernel_launch(void* const* d_in, const int* in_sizes, int n_in,
                              void* d_out, int out_size, void* d_ws, size_t ws_size,
                              hipStream_t stream) {
    const float* x        = (const float*)d_in[0];
    const int*   ei       = (const int*)d_in[1];
    const float* W1       = (const float*)d_in[2];
    const float* att_src1 = (const float*)d_in[3];
    const float* att_dst1 = (const float*)d_in[4];
    const float* b1       = (const float*)d_in[5];
    const float* W2       = (const float*)d_in[6];
    const float* att_src2 = (const float*)d_in[7];
    const float* att_dst2 = (const float*)d_in[8];
    const float* b2       = (const float*)d_in[9];
    const float* W3       = (const float*)d_in[10];
    const float* att_src3 = (const float*)d_in[11];
    const float* att_dst3 = (const float*)d_in[12];
    const float* b3       = (const float*)d_in[13];
    const float* Wc       = (const float*)d_in[14];
    const float* bc       = (const float*)d_in[15];

    const int Fin = 128, HC = 256, H = 4, OUT = 10;
    int N = in_sizes[0] / Fin;
    int E = in_sizes[1] / 2;

    // workspace carve-up
    char* ws = (char*)d_ws;
    size_t off = 0;
    auto alloc = [&](size_t bytes) -> void* {
        void* p = ws + off;
        off += (bytes + 255) & ~(size_t)255;
        return p;
    };
    unsigned short* act_bf = (unsigned short*)alloc((size_t)N * HC * sizeof(unsigned short));
    float* xw   = (float*)alloc((size_t)N * HC * sizeof(float));
    float* outb = (float*)alloc((size_t)N * HC * sizeof(float));
    float* a_src = (float*)alloc((size_t)N * H * sizeof(float));
    float* a_dst = (float*)alloc((size_t)N * H * sizeof(float));
    float* mbuf  = (float*)alloc((size_t)N * H * sizeof(float));
    float* sbuf  = (float*)alloc((size_t)N * H * sizeof(float));
    unsigned short* Wt = (unsigned short*)alloc((size_t)HC * HC * sizeof(unsigned short));
    float* g = (float*)alloc(64 * sizeof(float));

    // ---- layer 1: [N,128] -> [N,256], H=4, concat, ELU ----
    long long nx = (long long)N * Fin;
    cvt_bf16_kernel<<<(unsigned)((nx + 255) / 256), 256, 0, stream>>>(x, act_bf, nx);
    run_gat_layer(act_bf, Fin, HC, H, W1, att_src1, att_dst1, b1, /*elu=*/1,
                  ei, N, E, Wt, xw, outb, a_src, a_dst, mbuf, sbuf, stream);
    long long nh = (long long)N * HC;
    cvt_bf16_kernel<<<(unsigned)((nh + 255) / 256), 256, 0, stream>>>(outb, act_bf, nh);

    // ---- layer 2: [N,256] -> [N,256], H=4, concat, ELU ----
    run_gat_layer(act_bf, HC, HC, H, W2, att_src2, att_dst2, b2, /*elu=*/1,
                  ei, N, E, Wt, xw, outb, a_src, a_dst, mbuf, sbuf, stream);
    cvt_bf16_kernel<<<(unsigned)((nh + 255) / 256), 256, 0, stream>>>(outb, act_bf, nh);

    // ---- layer 3: [N,256] -> [N,64], H=1, mean(=identity), no ELU ----
    run_gat_layer(act_bf, HC, 64, 1, W3, att_src3, att_dst3, b3, /*elu=*/0,
                  ei, N, E, Wt, xw, outb, a_src, a_dst, mbuf, sbuf, stream);

    // ---- global mean pooling + classifier ----
    fill_kernel<<<1, 64, 0, stream>>>(g, 0.f, 64);
    colsum_kernel<<<512, 256, 0, stream>>>(outb, g, N);
    final_kernel<<<1, 32, 0, stream>>>(g, Wc, bc, (float*)d_out, N, OUT);
}
